// MowstStyleMoE_67783173865961
// MI455X (gfx1250) — matrix operand order repaired and checked by reading, not compiled, not run on hardware
//
#include <hip/hip_runtime.h>
#include <stdint.h>

#define NN 200000
#define NE 800000

typedef unsigned int u32;
typedef unsigned short u16;
typedef __attribute__((ext_vector_type(16))) __bf16 v16bf;
typedef __attribute__((ext_vector_type(8)))  float  v8f;

union BF16Frag { v16bf v; u32 u[8]; uint4 q[2]; };

__device__ __forceinline__ u16 f2bf(float f) {
  u32 u = __builtin_bit_cast(u32, f);
  u32 r = u + 0x7FFFu + ((u >> 16) & 1u);
  return (u16)(r >> 16);
}
__device__ __forceinline__ float bf2f(u16 h) {
  u32 u = ((u32)h) << 16;
  return __builtin_bit_cast(float, u);
}
__device__ __forceinline__ u32 pk2(float a, float b) {
  return (u32)f2bf(a) | ((u32)f2bf(b) << 16);
}

// ---------------------------------------------------------------------------
// Repack one 256x256 row-major f32 weight into WMMA B-fragment order (bf16).
// Fragment (ntile, t): lane l: n = ntile*16 + (l&15), half = l>>4,
//   u[v] packs (W[k][n], W[k+1][n]) with k = 32t + (v>>2)*16 + half*8 + 2*(v&3).
// Linear id == (((ntile*8 + t)*32 + lane)*8 + v).  32768 threads total.
// ---------------------------------------------------------------------------
__global__ void __launch_bounds__(256)
repack_w(const float* __restrict__ W, u32* __restrict__ P) {
  int id    = blockIdx.x * 256 + threadIdx.x;     // 0..32767
  int v     = id & 7;
  int lane  = (id >> 3) & 31;
  int t     = (id >> 8) & 7;
  int ntile = id >> 11;
  int n = ntile * 16 + (lane & 15);
  int k = t * 32 + (v >> 2) * 16 + (lane >> 4) * 8 + 2 * (v & 3);
  P[id] = pk2(W[k * 256 + n], W[(k + 1) * 256 + n]);
}

// ---------------------------------------------------------------------------
// h0 = relu(xs @ e1 + be1), xs = x[:,4:10].  One thread per (node, j).
// ---------------------------------------------------------------------------
__global__ void __launch_bounds__(256)
hidden0_kernel(const float* __restrict__ x, const float* __restrict__ e1,
               const float* __restrict__ be1, u16* __restrict__ out) {
  int id = blockIdx.x * 256 + threadIdx.x;   // N*256 threads exactly
  int node = id >> 8;
  int j = id & 255;
  const float* xr = x + (size_t)node * 10 + 4;
  float acc = be1[j];
#pragma unroll
  for (int i = 0; i < 6; ++i) acc += xr[i] * e1[i * 256 + j];
  out[id] = f2bf(fmaxf(acc, 0.f));
}

// ---------------------------------------------------------------------------
// weak = relu(xs@w1+b1) @ w2 + b2, fused over j.  One thread per node.
// ---------------------------------------------------------------------------
__global__ void __launch_bounds__(256)
weak_kernel(const float* __restrict__ x, const float* __restrict__ w1,
            const float* __restrict__ b1, const float* __restrict__ w2,
            const float* __restrict__ b2, float* __restrict__ weak) {
  int node = blockIdx.x * 256 + threadIdx.x;
  if (node >= NN) return;
  float xs[6];
#pragma unroll
  for (int i = 0; i < 6; ++i) xs[i] = x[(size_t)node * 10 + 4 + i];
  float a0 = b2[0], a1 = b2[1], a2 = b2[2], a3 = b2[3];
  for (int j = 0; j < 256; ++j) {
    float hj = b1[j];
#pragma unroll
    for (int i = 0; i < 6; ++i) hj += xs[i] * w1[i * 256 + j];
    hj = fmaxf(hj, 0.f);
    a0 += hj * w2[j * 4 + 0];
    a1 += hj * w2[j * 4 + 1];
    a2 += hj * w2[j * 4 + 2];
    a3 += hj * w2[j * 4 + 3];
  }
  float* o = weak + (size_t)node * 4;
  o[0] = a0; o[1] = a1; o[2] = a2; o[3] = a3;
}

// ---------------------------------------------------------------------------
// segment_sum scatter: one 32-lane wave per edge, 8 fp32 atomics per lane.
// h is bf16 (N x 256), agg is f32 (N x 256), zeroed beforehand.
// ---------------------------------------------------------------------------
__global__ void __launch_bounds__(256)
scatter_kernel(const u16* __restrict__ h, const int* __restrict__ src,
               const int* __restrict__ tgt, float* __restrict__ agg) {
  int gid  = blockIdx.x * 256 + threadIdx.x;
  int edge = gid >> 5;
  int lane = gid & 31;
  if (edge >= NE) return;
  int s = src[edge];
  int t = tgt[edge];
  const u16* hs = h + (size_t)s * 256;
  float* ar = agg + (size_t)t * 256;
#pragma unroll
  for (int i = 0; i < 8; ++i) {
    int f = lane + i * 32;
    float v = bf2f(hs[f]);
    float* p = ar + f;
    asm volatile("global_atomic_add_f32 %0, %1, off"
                 :: "v"(p), "v"(v) : "memory");
  }
}

// ---------------------------------------------------------------------------
// WMMA GEMM: D = [agg @ Bagg] + h @ Bh + bias, optional relu, bf16 output.
// agg: f32 (HAS_AGG), h: bf16, both N x 256 row-major. B* are repacked.
// One wave computes a 16x64 strip (4 consecutive 16x16 N-tiles).
// Per K-step: batch-load ALL fragments (one big clause), then run the
// WMMAs straight-line so partial load-count waits can overlap latency.
// 8 waves/block. waves = 12500 M-tiles * 4 groups = 50000 -> 6250 blocks.
// ---------------------------------------------------------------------------
template <bool HAS_AGG, bool RELU>
__global__ void __launch_bounds__(256)
conv_gemm(const float* __restrict__ agg, const u32* __restrict__ Bagg,
          const u16* __restrict__ h, const u32* __restrict__ Bh,
          const float* __restrict__ bias, u16* __restrict__ D) {
  const int wave   = blockIdx.x * 8 + (threadIdx.x >> 5);
  const int lane   = threadIdx.x & 31;
  const int tile_m = wave >> 2;          // 0..12499
  const int g      = wave & 3;           // N-group (ntiles 4g..4g+3)
  const int half   = lane >> 4;
  const int m      = lane & 15;

  const char*  hrow = (const char*)(h + (size_t)(tile_m * 16 + m) * 256);
  const float* arow = HAS_AGG ? (agg + (size_t)(tile_m * 16 + m) * 256) : nullptr;

  // Per-lane fragment bases into the packed weights (contiguous 32B per lane).
  const u32* bhbase = Bh + (((size_t)(g * 4) * 8) * 32 + lane) * 8;
  const u32* babase = HAS_AGG ? (Bagg + (((size_t)(g * 4) * 8) * 32 + lane) * 8)
                              : nullptr;

  v8f acc[4] = {};

  for (int t = 0; t < 8; ++t) {
    // ---- batch all loads for this K-step ----
    BF16Frag ah;
    ah.q[0] = *(const uint4*)(hrow + t * 64 + half * 16);
    ah.q[1] = *(const uint4*)(hrow + t * 64 + half * 16 + 32);

    BF16Frag bh[4];
#pragma unroll
    for (int j = 0; j < 4; ++j) {
      const u32* bp = bhbase + ((size_t)j * 8 + t) * 256;   // ntile stride = 2048 u32
      bh[j].q[0] = *(const uint4*)(bp);
      bh[j].q[1] = *(const uint4*)(bp + 4);
    }

    BF16Frag aa;
    BF16Frag ba[4];
    if (HAS_AGG) {
      const float* ab = arow + t * 32 + half * 8;
      float4 f0 = *(const float4*)(ab);
      float4 f1 = *(const float4*)(ab + 4);
      float4 f2 = *(const float4*)(ab + 16);
      float4 f3 = *(const float4*)(ab + 20);
#pragma unroll
      for (int j = 0; j < 4; ++j) {
        const u32* bp = babase + ((size_t)j * 8 + t) * 256;
        ba[j].q[0] = *(const uint4*)(bp);
        ba[j].q[1] = *(const uint4*)(bp + 4);
      }
      aa.u[0] = pk2(f0.x, f0.y); aa.u[1] = pk2(f0.z, f0.w);
      aa.u[2] = pk2(f1.x, f1.y); aa.u[3] = pk2(f1.z, f1.w);
      aa.u[4] = pk2(f2.x, f2.y); aa.u[5] = pk2(f2.z, f2.w);
      aa.u[6] = pk2(f3.x, f3.y); aa.u[7] = pk2(f3.z, f3.w);
    }

    // ---- straight-line WMMAs ----
#pragma unroll
    for (int j = 0; j < 4; ++j) {
      acc[j] = __builtin_amdgcn_wmma_f32_16x16x32_bf16(
          false, ah.v, false, bh[j].v, (short)0, acc[j], false, false);
      if (HAS_AGG) {
        acc[j] = __builtin_amdgcn_wmma_f32_16x16x32_bf16(
            false, aa.v, false, ba[j].v, (short)0, acc[j], false, false);
      }
    }
  }

  // Epilogue: C/D layout -> row = tile_m*16 + half*8 + r, col = ntile*16 + (lane&15)
  const int col0 = lane & 15;
#pragma unroll
  for (int j = 0; j < 4; ++j) {
    const int col = (g * 4 + j) * 16 + col0;
    const float b = bias[col];
#pragma unroll
    for (int r = 0; r < 8; ++r) {
      float v = acc[j][r] + b;
      if (RELU) v = fmaxf(v, 0.f);
      const int row = tile_m * 16 + half * 8 + r;
      D[(size_t)row * 256 + col] = f2bf(v);
    }
  }
}

// ---------------------------------------------------------------------------
// Final layer (OUT=4): strong = agg@rw3 + h@sw3 + rb3, then conf-blend.
// One wave per node with shuffle reduction; lane 0 does softmax/conf/blend.
// ---------------------------------------------------------------------------
__global__ void __launch_bounds__(256)
final_kernel(const float* __restrict__ agg, const u16* __restrict__ h,
             const float* __restrict__ rw3, const float* __restrict__ sw3,
             const float* __restrict__ rb3, const float* __restrict__ weak,
             float* __restrict__ out) {
  int gid  = blockIdx.x * 256 + threadIdx.x;
  int node = gid >> 5;
  int lane = gid & 31;
  if (node >= NN) return;
  const float* ar = agg + (size_t)node * 256;
  const u16*   hr = h   + (size_t)node * 256;
  float a[4] = {0.f, 0.f, 0.f, 0.f};
#pragma unroll
  for (int i = 0; i < 8; ++i) {
    int k = lane + i * 32;
    float av = ar[k];
    float hv = bf2f(hr[k]);
#pragma unroll
    for (int o = 0; o < 4; ++o)
      a[o] += av * rw3[k * 4 + o] + hv * sw3[k * 4 + o];
  }
#pragma unroll
  for (int o = 0; o < 4; ++o)
    for (int off = 16; off > 0; off >>= 1)
      a[o] += __shfl_down(a[o], off, 32);

  if (lane == 0) {
    float w[4], strong[4];
#pragma unroll
    for (int o = 0; o < 4; ++o) {
      strong[o] = a[o] + rb3[o];
      w[o] = weak[(size_t)node * 4 + o];
    }
    float mx = fmaxf(fmaxf(w[0], w[1]), fmaxf(w[2], w[3]));
    float e[4], s = 0.f;
#pragma unroll
    for (int o = 0; o < 4; ++o) { e[o] = __expf(w[o] - mx); s += e[o]; }
    float inv = 1.f / s;
    float p2 = 0.f, ent = 0.f;
#pragma unroll
    for (int o = 0; o < 4; ++o) {
      float p = e[o] * inv;
      p2 += p * p;
      ent -= p * __logf(p + 1e-8f);
    }
    float var = p2 * 0.25f - 0.0625f;          // var over 4 probs (mean = 0.25)
    ent = ent * (1.f / 1.3862943611f);          // / log(4)
    float conf = 0.5f * (var + (1.f - ent));
    conf = fminf(fmaxf(conf, 0.f), 1.f);
#pragma unroll
    for (int o = 0; o < 4; ++o)
      out[(size_t)node * 4 + o] = conf * w[o] + (1.f - conf) * strong[o];
  }
}

// ---------------------------------------------------------------------------
// Launch sequence.
// Workspace layout (bytes):
//   [0,            102.4e6) hA  bf16 N*256
//   [102.4e6,      204.8e6) hB  bf16 N*256
//   [204.8e6,      409.6e6) agg f32  N*256
//   [409.6e6,      412.8e6) weak f32 N*4
//   [412.8e6,      ~413.7e6) 7 packed weights, 131072 B each
// ---------------------------------------------------------------------------
extern "C" void kernel_launch(void* const* d_in, const int* in_sizes, int n_in,
                              void* d_out, int out_size, void* d_ws, size_t ws_size,
                              hipStream_t stream) {
  const float* x    = (const float*)d_in[0];
  const int*   ei   = (const int*)d_in[1];      // int64 in ref demotes to i32 (jax x64 off)
  const float* w1   = (const float*)d_in[2];
  const float* b1   = (const float*)d_in[3];
  const float* w2   = (const float*)d_in[4];
  const float* b2   = (const float*)d_in[5];
  const float* e1   = (const float*)d_in[6];
  const float* be1  = (const float*)d_in[7];
  const float* e2   = (const float*)d_in[8];
  const float* be2  = (const float*)d_in[9];
  const float* rw0  = (const float*)d_in[10];
  const float* rb0  = (const float*)d_in[11];
  const float* sw0  = (const float*)d_in[12];
  const float* rw1  = (const float*)d_in[13];
  const float* rb1  = (const float*)d_in[14];
  const float* sw1  = (const float*)d_in[15];
  const float* rw2  = (const float*)d_in[16];
  const float* rb2  = (const float*)d_in[17];
  const float* sw2  = (const float*)d_in[18];
  const float* rw3  = (const float*)d_in[19];
  const float* rb3  = (const float*)d_in[20];
  const float* sw3  = (const float*)d_in[21];

  char* ws = (char*)d_ws;
  u16*   hA    = (u16*)ws;
  u16*   hB    = (u16*)(ws + 102400000ull);
  float* agg   = (float*)(ws + 204800000ull);
  float* weak  = (float*)(ws + 409600000ull);
  u32*   packW = (u32*)(ws + 412800000ull);
  const size_t WSLOT = 32768;   // u32 per packed weight

  const int* src = ei;
  const int* tgt = ei + NE;

  // 1) repack weights to WMMA B-fragment order (bf16)
  repack_w<<<128, 256, 0, stream>>>(e2,  packW + 0 * WSLOT);
  repack_w<<<128, 256, 0, stream>>>(rw0, packW + 1 * WSLOT);
  repack_w<<<128, 256, 0, stream>>>(sw0, packW + 2 * WSLOT);
  repack_w<<<128, 256, 0, stream>>>(rw1, packW + 3 * WSLOT);
  repack_w<<<128, 256, 0, stream>>>(sw1, packW + 4 * WSLOT);
  repack_w<<<128, 256, 0, stream>>>(rw2, packW + 5 * WSLOT);
  repack_w<<<128, 256, 0, stream>>>(sw2, packW + 6 * WSLOT);

  // 2) small-K node MLPs
  hidden0_kernel<<<NN, 256, 0, stream>>>(x, e1, be1, hB);            // h0 -> hB
  weak_kernel<<<(NN + 255) / 256, 256, 0, stream>>>(x, w1, b1, w2, b2, weak);

  // 3) h = h0 @ e2 + be2   (WMMA, no relu) -> hA
  conv_gemm<false, false><<<6250, 256, 0, stream>>>(
      nullptr, nullptr, hB, packW + 0 * WSLOT, be2, hA);

  // 4) three graph-conv layers: agg = segsum(h); h = relu(agg@rw + h@sw + rb)
  hipMemsetAsync(agg, 0, (size_t)NN * 256 * 4, stream);
  scatter_kernel<<<100000, 256, 0, stream>>>(hA, src, tgt, agg);
  conv_gemm<true, true><<<6250, 256, 0, stream>>>(
      agg, packW + 1 * WSLOT, hA, packW + 2 * WSLOT, rb0, hB);

  hipMemsetAsync(agg, 0, (size_t)NN * 256 * 4, stream);
  scatter_kernel<<<100000, 256, 0, stream>>>(hB, src, tgt, agg);
  conv_gemm<true, true><<<6250, 256, 0, stream>>>(
      agg, packW + 3 * WSLOT, hB, packW + 4 * WSLOT, rb1, hA);

  hipMemsetAsync(agg, 0, (size_t)NN * 256 * 4, stream);
  scatter_kernel<<<100000, 256, 0, stream>>>(hA, src, tgt, agg);
  conv_gemm<true, true><<<6250, 256, 0, stream>>>(
      agg, packW + 5 * WSLOT, hA, packW + 6 * WSLOT, rb2, hB);

  // 5) layer 3 (OUT=4) + confidence blend -> d_out
  hipMemsetAsync(agg, 0, (size_t)NN * 256 * 4, stream);
  scatter_kernel<<<100000, 256, 0, stream>>>(hB, src, tgt, agg);
  final_kernel<<<25000, 256, 0, stream>>>(agg, hB, rw3, sw3, rb3, weak,
                                          (float*)d_out);
}
